// MultiScaleRetention_62543313764845
// MI455X (gfx1250) — matrix operand order repaired and checked by reading, not compiled, not run on hardware
//
#include <hip/hip_runtime.h>
#include <cstdint>
#include <cstddef>

typedef __attribute__((ext_vector_type(16))) _Float16 v16h;
typedef __attribute__((ext_vector_type(8)))  _Float16 h8v;
typedef __attribute__((ext_vector_type(8)))  float    v8f;

#define B_   8
#define T_   2048
#define D_   1024
#define H_   16
#define HD_  64
#define SC_  16
#define M_   (B_ * T_)        /* 16384 tokens */
#define NCHUNK 8
#define TCHUNK (T_ / NCHUNK)  /* 256 steps per chunk */

#define LDS_LD 40             /* padded row stride in halves (32 data + 8 pad) */

// ---------------------------------------------------------------- helpers
__device__ __forceinline__ float sigmoidf_(float v) {
  return 1.0f / (1.0f + __expf(-v));
}

// raw LDS byte offset of a __shared__ pointer (generic -> addrspace(3) -> int)
__device__ __forceinline__ uint32_t lds_off_u32(const void* p) {
  return (uint32_t)(uintptr_t)(const __attribute__((address_space(3))) char*)p;
}

// gfx1250 async DMA: 16B global -> LDS, tracked by ASYNCcnt
__device__ __forceinline__ void async_g2l_b128(uint32_t lds_byte_off,
                                               const _Float16* gptr) {
  asm volatile("global_load_async_to_lds_b128 %0, %1, off"
               :: "v"(lds_byte_off), "v"(gptr)
               : "memory");
}

__device__ __forceinline__ void wait_asynccnt0() {
  asm volatile("s_wait_asynccnt 0x0" ::: "memory");
}

// ---------------------------------------------------------------- convert
__global__ void cvt_f32_f16_kernel(const float* __restrict__ src,
                                   _Float16* __restrict__ dst, int n) {
  int i = blockIdx.x * blockDim.x + threadIdx.x;
  int stride = gridDim.x * blockDim.x;
  for (; i < n; i += stride) dst[i] = (_Float16)src[i];
}

// ---------------------------------------------------------------- WMMA GEMM
// C[M,N] = A[M,K] (f16, row-major) * Bw[N,K]^T (f16, row-major)
// Block: 128 threads = 4 waves (2x2), 128x128 block tile, wave tile 64x64.
// A/B K-tiles (128x32 each) staged to LDS with async DMA, double buffered.
// Per K-step per wave: 16 ds_load_b128 -> 16 v_wmma (A,B frags reused 4x).
// MODE 0: C[m,n] = acc                         (h = x @ W_in^T)
// MODE 1: g = sigmoid(acc + bias[n]);          (gate fused epilogue)
//         yb[m,n] = (f16)(g*y0 + (1-g)*x)
// MODE 2: C[m,n] = acc                         (out = y @ W_out^T)
template <int MODE>
__global__ __launch_bounds__(128) void gemm_wmma_f16(
    const _Float16* __restrict__ A, const _Float16* __restrict__ Bw,
    float* __restrict__ C, const float* __restrict__ bias,
    const float* __restrict__ xf, const float* __restrict__ y0,
    _Float16* __restrict__ yb, int M, int N, int K)
{
  __shared__ _Float16 ldsA[2][128 * LDS_LD];
  __shared__ _Float16 ldsB[2][128 * LDS_LD];

  const int tid  = threadIdx.x;
  const int lane = tid & 31;
  const int w    = tid >> 5;       // wave 0..3
  const int wm   = w >> 1;         // 0..1  (M direction)
  const int wn   = w & 1;          // 0..1  (N direction)
  const int half = lane >> 4;      // 0/1
  const int l16  = lane & 15;

  const int rowA = blockIdx.y * 128;            // block M origin
  const int colB = blockIdx.x * 128;            // block N origin
  const int row0 = wm * 64;                     // wave tile origin in block
  const int col0 = wn * 64;

  // stage one 128x32 A tile + 128x32 B tile into LDS buffer `buf`
  // (128 threads x 8 async b128 = 16KB per K-step; one row per thread)
  const int srow = tid;                         // 0..127
  auto stage = [&](int buf, int k0) {
    const _Float16* ga = A  + (size_t)(rowA + srow) * K + k0;
    const _Float16* gb = Bw + (size_t)(colB + srow) * K + k0;
    uint32_t la = lds_off_u32(&ldsA[buf][srow * LDS_LD]);
    uint32_t lb = lds_off_u32(&ldsB[buf][srow * LDS_LD]);
#pragma unroll
    for (int q = 0; q < 4; ++q) {
      async_g2l_b128(la + q * 16, ga + q * 8);
      async_g2l_b128(lb + q * 16, gb + q * 8);
    }
  };

  v8f zero = {};
  v8f acc[4][4];
#pragma unroll
  for (int i = 0; i < 4; ++i)
#pragma unroll
    for (int j = 0; j < 4; ++j) acc[i][j] = zero;

  stage(0, 0);

  int buf = 0;
  for (int k0 = 0; k0 < K; k0 += 32, buf ^= 1) {
    wait_asynccnt0();        // my async writes for `buf` have landed in LDS
    __syncthreads();         // everyone's have; everyone done reading buf^1

    if (k0 + 32 < K) stage(buf ^ 1, k0 + 32);  // overlap DMA with WMMA

    // B fragments: lane col n = l16; elems 0..7 -> K = 8*half + e,
    // elems 8..15 -> K = 16 + 8*half + (e-8)
    v16h bfr[4];
#pragma unroll
    for (int j = 0; j < 4; ++j) {
      const _Float16* p = &ldsB[buf][(col0 + j * 16 + l16) * LDS_LD + half * 8];
      h8v lo = *(const h8v*)p;
      h8v hi = *(const h8v*)(p + 16);
#pragma unroll
      for (int e = 0; e < 8; ++e) { bfr[j][e] = lo[e]; bfr[j][e + 8] = hi[e]; }
    }
    // A fragments: lane m = l16; elems 0..7 -> K = 8*half + e,
    // elems 8..15 -> K = 16 + 8*half + (e-8)   (ISA 16-bit A layout)
#pragma unroll
    for (int i = 0; i < 4; ++i) {
      const _Float16* p = &ldsA[buf][(row0 + i * 16 + l16) * LDS_LD + half * 8];
      h8v lo = *(const h8v*)p;
      h8v hi = *(const h8v*)(p + 16);
      v16h af;
#pragma unroll
      for (int e = 0; e < 8; ++e) { af[e] = lo[e]; af[e + 8] = hi[e]; }
#pragma unroll
      for (int j = 0; j < 4; ++j)
        acc[i][j] = __builtin_amdgcn_wmma_f32_16x16x32_f16(
            false, af, false, bfr[j], (short)0, acc[i][j], false, false);
    }
  }

  // C/D layout: VGPR r -> M = r + 8*half, N = l16
#pragma unroll
  for (int i = 0; i < 4; ++i)
#pragma unroll
    for (int j = 0; j < 4; ++j)
#pragma unroll
      for (int r = 0; r < 8; ++r) {
        const int m = rowA + row0 + i * 16 + half * 8 + r;
        const int n = colB + col0 + j * 16 + l16;
        const size_t idx = (size_t)m * N + n;
        const float v = acc[i][j][r];
        if (MODE == 1) {
          const float g = sigmoidf_(v + bias[n]);
          yb[idx] = (_Float16)(g * y0[idx] + (1.0f - g) * xf[idx]);
        } else {
          C[idx] = v;
        }
      }
}

// ---------------------------------------------------------------- scan phase 1
// Per (chunk c, b, h, d): local scan of 256 steps from S=0; store final S[s].
// Thread with (h==0 && d==0) also records prod(keep) for its (c,b).
__global__ void scan_chunk_local(const float* __restrict__ hbuf,
                                 const unsigned char* __restrict__ pad,
                                 const float* __restrict__ logit,
                                 float* __restrict__ chunkS,
                                 float* __restrict__ keepprod)
{
  const int id = blockIdx.x * blockDim.x + threadIdx.x;  // < NCHUNK*B_*H_*HD_
  const int d = id & (HD_ - 1);
  const int h = (id >> 6) & (H_ - 1);
  const int b = (id >> 10) & (B_ - 1);
  const int c = id >> 13;

  float dec[SC_], S[SC_];
#pragma unroll
  for (int s = 0; s < SC_; ++s) {
    dec[s] = sigmoidf_(logit[h * SC_ + s]);
    S[s] = 0.0f;
  }
  float kp = 1.0f;
  const int tbase = c * TCHUNK;
  for (int tl = 0; tl < TCHUNK; ++tl) {
    const int t = tbase + tl;
    const float keep = pad[b * T_ + t] ? 0.0f : 1.0f;
    const float ht = hbuf[(size_t)(b * T_ + t) * D_ + h * HD_ + d];
    kp *= keep;
#pragma unroll
    for (int s = 0; s < SC_; ++s) S[s] = (dec[s] * S[s] + ht) * keep;
  }
  const size_t base = (size_t)((c * B_ + b) * H_ + h) * (SC_ * HD_) + d;
#pragma unroll
  for (int s = 0; s < SC_; ++s) chunkS[base + s * HD_] = S[s];
  if (h == 0 && d == 0) keepprod[c * B_ + b] = kp;
}

// ---------------------------------------------------------------- scan phase 2
// Per (b,h,s,d): sequential prefix over 8 chunks.
// carry_in[c] = A_{c-1}*carry_in[c-1] + B_{c-1},  A_c = decay^256 * prod(keep)
__global__ void scan_prefix(const float* __restrict__ chunkS,
                            const float* __restrict__ keepprod,
                            const float* __restrict__ logit,
                            float* __restrict__ carry)
{
  const int id = blockIdx.x * blockDim.x + threadIdx.x;  // < B_*H_*SC_*HD_
  const int d = id & (HD_ - 1);
  const int s = (id >> 6) & (SC_ - 1);
  const int h = (id >> 10) & (H_ - 1);
  const int b = (id >> 14) & (B_ - 1);

  const float dec = sigmoidf_(logit[h * SC_ + s]);
  float Abase = dec;
#pragma unroll
  for (int i = 0; i < 8; ++i) Abase *= Abase;  // dec^256

  float cr = 0.0f;
  for (int c = 0; c < NCHUNK; ++c) {
    const size_t idx = (size_t)((c * B_ + b) * H_ + h) * (SC_ * HD_) + (size_t)s * HD_ + d;
    carry[idx] = cr;
    cr = Abase * keepprod[c * B_ + b] * cr + chunkS[idx];
  }
}

// ---------------------------------------------------------------- scan phase 3
// Replay chunks from corrected initial state; emit y0 = einsum(S, state_mix).
__global__ void scan_emit(const float* __restrict__ hbuf,
                          const unsigned char* __restrict__ pad,
                          const float* __restrict__ logit,
                          const float* __restrict__ mixw,
                          const float* __restrict__ carry,
                          float* __restrict__ y0)
{
  const int id = blockIdx.x * blockDim.x + threadIdx.x;  // < NCHUNK*B_*H_*HD_
  const int d = id & (HD_ - 1);
  const int h = (id >> 6) & (H_ - 1);
  const int b = (id >> 10) & (B_ - 1);
  const int c = id >> 13;

  float dec[SC_], S[SC_], mx[SC_];
  const size_t cbase = (size_t)((c * B_ + b) * H_ + h) * (SC_ * HD_) + d;
#pragma unroll
  for (int s = 0; s < SC_; ++s) {
    dec[s] = sigmoidf_(logit[h * SC_ + s]);
    S[s] = carry[cbase + s * HD_];
    mx[s] = mixw[(h * SC_ + s) * HD_ + d];
  }
  const int tbase = c * TCHUNK;
  for (int tl = 0; tl < TCHUNK; ++tl) {
    const int t = tbase + tl;
    const float keep = pad[b * T_ + t] ? 0.0f : 1.0f;
    const float ht = hbuf[(size_t)(b * T_ + t) * D_ + h * HD_ + d];
    float r = 0.0f;
#pragma unroll
    for (int s = 0; s < SC_; ++s) {
      S[s] = (dec[s] * S[s] + ht) * keep;
      r += mx[s] * S[s];
    }
    y0[(size_t)(b * T_ + t) * D_ + h * HD_ + d] = r;
  }
}

// ---------------------------------------------------------------- launch
extern "C" void kernel_launch(void* const* d_in, const int* in_sizes, int n_in,
                              void* d_out, int out_size, void* d_ws, size_t ws_size,
                              hipStream_t stream)
{
  (void)in_sizes; (void)n_in; (void)out_size; (void)ws_size;

  const float*         x      = (const float*)d_in[0];
  const unsigned char* pad    = (const unsigned char*)d_in[1];
  const float*         W_in   = (const float*)d_in[2];
  const float*         mixw   = (const float*)d_in[3];
  const float*         logit  = (const float*)d_in[4];
  const float*         W_gate = (const float*)d_in[5];
  const float*         b_gate = (const float*)d_in[6];
  const float*         W_out  = (const float*)d_in[7];
  float*               out    = (float*)d_out;

  // workspace layout
  char* ws = (char*)d_ws;
  size_t o = 0;
  _Float16* xb    = (_Float16*)(ws + o); o += (size_t)M_ * D_ * sizeof(_Float16);
  _Float16* wbin  = (_Float16*)(ws + o); o += (size_t)D_ * D_ * sizeof(_Float16);
  _Float16* wbg   = (_Float16*)(ws + o); o += (size_t)D_ * D_ * sizeof(_Float16);
  _Float16* wbo   = (_Float16*)(ws + o); o += (size_t)D_ * D_ * sizeof(_Float16);
  float*    hbuf  = (float*)(ws + o);    o += (size_t)M_ * D_ * sizeof(float);
  float*    y0    = (float*)(ws + o);    o += (size_t)M_ * D_ * sizeof(float);
  _Float16* yb    = (_Float16*)(ws + o); o += (size_t)M_ * D_ * sizeof(_Float16);
  float*    chkS  = (float*)(ws + o);    o += (size_t)NCHUNK * B_ * H_ * SC_ * HD_ * sizeof(float);
  float*    carry = (float*)(ws + o);    o += (size_t)NCHUNK * B_ * H_ * SC_ * HD_ * sizeof(float);
  float*    kprod = (float*)(ws + o);    o += 256;

  // 1. f32 -> f16 conversions
  cvt_f32_f16_kernel<<<4096, 256, 0, stream>>>(x, xb, M_ * D_);
  cvt_f32_f16_kernel<<<1024, 256, 0, stream>>>(W_in, wbin, D_ * D_);
  cvt_f32_f16_kernel<<<1024, 256, 0, stream>>>(W_gate, wbg, D_ * D_);
  cvt_f32_f16_kernel<<<1024, 256, 0, stream>>>(W_out, wbo, D_ * D_);

  dim3 ggrid(D_ / 128, M_ / 128);  // (8, 128)

  // 2. h = x @ W_in^T
  gemm_wmma_f16<0><<<ggrid, 128, 0, stream>>>(xb, wbin, hbuf, nullptr,
                                              nullptr, nullptr, nullptr,
                                              M_, D_, D_);

  // 3. chunked parallel scan
  scan_chunk_local<<<(NCHUNK * B_ * H_ * HD_) / 256, 256, 0, stream>>>(
      hbuf, pad, logit, chkS, kprod);
  scan_prefix<<<(B_ * H_ * SC_ * HD_) / 256, 256, 0, stream>>>(
      chkS, kprod, logit, carry);
  scan_emit<<<(NCHUNK * B_ * H_ * HD_) / 256, 256, 0, stream>>>(
      hbuf, pad, logit, mixw, carry, y0);

  // 4. gate GEMM with fused sigmoid/mix epilogue -> yb (f16)
  gemm_wmma_f16<1><<<ggrid, 128, 0, stream>>>(xb, wbg, nullptr, b_gate,
                                              x, y0, yb,
                                              M_, D_, D_);

  // 5. out = y @ W_out^T
  gemm_wmma_f16<2><<<ggrid, 128, 0, stream>>>(yb, wbo, out, nullptr,
                                              nullptr, nullptr, nullptr,
                                              M_, D_, D_);
}
